// Net_66803921322270
// MI455X (gfx1250) — matrix operand order repaired and checked by reading, not compiled, not run on hardware
//
#include <hip/hip_runtime.h>

#define CCE const float* __restrict__

typedef __attribute__((ext_vector_type(16))) __bf16         v16bf;
typedef __attribute__((ext_vector_type(16))) unsigned short v16us;
typedef __attribute__((ext_vector_type(8)))  unsigned short v8us;
typedef __attribute__((ext_vector_type(8)))  float          v8f;

static __device__ __forceinline__ unsigned short f2bf(float x) {
    unsigned u = __float_as_uint(x);
    return (unsigned short)((u + 0x7FFFu + ((u >> 16) & 1u)) >> 16);
}
static inline int ceil32(int x) { return (x + 31) & ~31; }

// A fragment: lane holds K {off..off+7, off+16..off+23}: two contiguous b128s
static __device__ __forceinline__ v16bf ld_a_frag(const unsigned short* p) {
    v8us lo = *(const v8us*)p;
    v8us hi = *(const v8us*)(p + 16);
    return __builtin_bit_cast(
        v16bf, __builtin_shufflevector(lo, hi, 0, 1, 2, 3, 4, 5, 6, 7,
                                       8, 9, 10, 11, 12, 13, 14, 15));
}
// B fragment: lane holds K {off..off+7, off+16..off+23}: two contiguous b128s
static __device__ __forceinline__ v16bf ld_b_frag(const unsigned short* p) {
    v8us lo = *(const v8us*)p;
    v8us hi = *(const v8us*)(p + 16);
    return __builtin_bit_cast(
        v16bf, __builtin_shufflevector(lo, hi, 0, 1, 2, 3, 4, 5, 6, 7,
                                       8, 9, 10, 11, 12, 13, 14, 15));
}
static __device__ __forceinline__ v8f wmma_bf16(v16bf a, v16bf b, v8f c) {
    return __builtin_amdgcn_wmma_f32_16x16x32_bf16(false, a, false, b,
                                                   (short)0, c, false, false);
}

// ---------------------------------------------------------------------------
// Fold eval-mode BatchNorm into linear weights, store TRANSPOSED + padded:
//   WT[o, k] ([Np][Kp] bf16) = W[k, o] * g[o]*rsqrt(v[o]+eps); zero in pads.
// ---------------------------------------------------------------------------
__global__ void fold_kernel(CCE W, CCE b, CCE g, CCE bb, CCE m, CCE v,
                            unsigned short* __restrict__ WT, float* __restrict__ bout,
                            int ci, int co, int Kp, int Np, int hasbn) {
    int t = blockIdx.x * blockDim.x + threadIdx.x;
    if (t < Np * Kp) {
        int k = t % Kp, o = t / Kp;
        float val = 0.0f;
        if (o < co && k < ci) {
            float s = hasbn ? g[o] * rsqrtf(v[o] + 1e-5f) : 1.0f;
            val = W[(size_t)k * co + o] * s;
        }
        WT[t] = f2bf(val);
    }
    if (t < Np) {
        float bv = 0.0f;
        if (t < co) {
            float s  = hasbn ? g[t] * rsqrtf(v[t] + 1e-5f) : 1.0f;
            float mm = hasbn ? m[t] : 0.0f;
            float be = hasbn ? bb[t] : 0.0f;
            bv = (b[t] - mm) * s + be;
        }
        bout[t] = bv;
    }
}

// ---------------------------------------------------------------------------
// WMMA bf16 GEMM with 2x2 tile register blocking: one wave owns a 32x32
// output block (M % 32 == 0 always here; Np % 32 == 0 except the padded
// 16-wide classifier tail, handled by the wave-uniform n2 guard).
// Per 32-K step: 8x global_load_b128 -> 4x v_wmma (16 FLOP/byte).
// EXEC stays all-ones at every WMMA (all guards wave-uniform).
// ---------------------------------------------------------------------------
template <bool RELU, bool ADD, bool OUT16>
__global__ __launch_bounds__(256) void gemm_wmma_kernel(
    const unsigned short* __restrict__ A, const unsigned short* __restrict__ WT,
    CCE bias, CCE Add, void* __restrict__ Yv, int M, int Kp, int Np) {
    const int lane    = threadIdx.x & 31;
    const int wave    = threadIdx.x >> 5;
    const int tilesN2 = (Np + 31) >> 5;
    const int tile    = blockIdx.x * 8 + wave;           // wave-uniform
    if (tile >= (M >> 5) * tilesN2) return;              // uniform exit
    const int tm   = (tile / tilesN2) << 5;
    const int tn   = (tile % tilesN2) << 5;
    const int half = lane >> 4;
    const int l16  = lane & 15;
    const bool n2  = (tn + 16) < Np;                     // uniform

    const unsigned short* pa0 = A  + (size_t)(tm + l16) * Kp + (half << 3);
    const unsigned short* pa1 = pa0 + (size_t)16 * Kp;
    const unsigned short* pb0 = WT + (size_t)(tn + l16) * Kp + (half << 3);
    const unsigned short* pb1 = pb0 + (size_t)16 * Kp;   // touched only if n2

    v8f c00 = {}, c01 = {}, c10 = {}, c11 = {};
    if (n2) {
        for (int kk = 0; kk < Kp; kk += 32) {
            v16bf a0 = ld_a_frag(pa0), a1 = ld_a_frag(pa1);
            v16bf b0 = ld_b_frag(pb0), b1 = ld_b_frag(pb1);
            c00 = wmma_bf16(a0, b0, c00);
            c01 = wmma_bf16(a0, b1, c01);
            c10 = wmma_bf16(a1, b0, c10);
            c11 = wmma_bf16(a1, b1, c11);
            pa0 += 32; pa1 += 32; pb0 += 32; pb1 += 32;
        }
    } else {
        for (int kk = 0; kk < Kp; kk += 32) {
            v16bf a0 = ld_a_frag(pa0), a1 = ld_a_frag(pa1);
            v16bf b0 = ld_b_frag(pb0);
            c00 = wmma_bf16(a0, b0, c00);
            c10 = wmma_bf16(a1, b0, c10);
            pa0 += 32; pa1 += 32; pb0 += 32;
        }
    }

    auto store_tile = [&](v8f acc, int rtm, int col) {
        float bv = bias[col];
#pragma unroll
        for (int r = 0; r < 8; ++r) {
            int row  = rtm + r + (half << 3);
            float vv = acc[r] + bv;
            if (ADD)  vv += Add[(size_t)row * Np + col];
            if (RELU) vv = fmaxf(vv, 0.0f);
            if (OUT16) ((unsigned short*)Yv)[(size_t)row * Np + col] = f2bf(vv);
            else       ((float*)Yv)[(size_t)row * Np + col] = vv;
        }
    };
    store_tile(c00, tm,      tn + l16);
    store_tile(c10, tm + 16, tn + l16);
    if (n2) {
        store_tile(c01, tm,      tn + 16 + l16);
        store_tile(c11, tm + 16, tn + 16 + l16);
    }
}

// ---------------------------------------------------------------------------
// Farthest point sampling: one block per batch, LDS argmax reduction.
// ---------------------------------------------------------------------------
__global__ __launch_bounds__(256) void fps_kernel(CCE POS, int n, int np,
                                                  int* __restrict__ idx,
                                                  float* __restrict__ dist) {
    int b = blockIdx.x;
    const float* P = POS + (size_t)b * n * 3;
    int*   id = idx  + (size_t)b * np;
    float* D  = dist + (size_t)b * n;
    int tid = threadIdx.x;
    for (int i = tid; i < n; i += 256) D[i] = 3.4e38f;
    if (tid == 0) id[0] = 0;
    __shared__ float sv[256];
    __shared__ int   si[256];
    int last = 0;
    __syncthreads();
    for (int it = 1; it < np; ++it) {
        float lx = P[last * 3], ly = P[last * 3 + 1], lz = P[last * 3 + 2];
        float best = -1.0f; int bi = 0;
        for (int i = tid; i < n; i += 256) {
            float dx = P[i * 3] - lx, dy = P[i * 3 + 1] - ly, dz = P[i * 3 + 2] - lz;
            float d  = dx * dx + dy * dy + dz * dz;
            float nd = fminf(D[i], d);
            D[i] = nd;
            if (nd > best) { best = nd; bi = i; }
        }
        sv[tid] = best; si[tid] = bi;
        __syncthreads();
        for (int s = 128; s > 0; s >>= 1) {
            if (tid < s) {
                if (sv[tid + s] > sv[tid] ||
                    (sv[tid + s] == sv[tid] && si[tid + s] < si[tid])) {
                    sv[tid] = sv[tid + s]; si[tid] = si[tid + s];
                }
            }
            __syncthreads();
        }
        last = si[0];
        if (tid == 0) id[it] = last;
        __syncthreads();
    }
}

__global__ void gather_center_kernel(CCE POS, const int* __restrict__ idx,
                                     float* __restrict__ CPOS, int np, int n, int Btot) {
    int t = blockIdx.x * blockDim.x + threadIdx.x;
    if (t >= Btot * np * 3) return;
    int c = t % 3;
    int r = t / 3;
    int b = r / np;
    CPOS[t] = POS[((size_t)b * n + idx[r]) * 3 + c];
}

// kNN (Ksel nearest, insertion sort) + dilated random subsample of K.
__global__ void knn_kernel(CCE CPOS, CCE POS, int np, int n, int Ksel, int dil,
                           int K, unsigned seed, int* __restrict__ nb, int Btot) {
    int t = blockIdx.x * blockDim.x + threadIdx.x;
    if (t >= Btot * np) return;
    int b = t / np;
    float cx = CPOS[t * 3], cy = CPOS[t * 3 + 1], cz = CPOS[t * 3 + 2];
    float bd[48]; int bi[48];
    for (int i = 0; i < Ksel; ++i) { bd[i] = 3.4e38f; bi[i] = 0; }
    const float* P = POS + (size_t)b * n * 3;
    for (int j = 0; j < n; ++j) {
        float dx = P[j * 3] - cx, dy = P[j * 3 + 1] - cy, dz = P[j * 3 + 2] - cz;
        float d = dx * dx + dy * dy + dz * dz;
        if (d < bd[Ksel - 1]) {
            int q = Ksel - 1;
            while (q > 0 && bd[q - 1] > d) { bd[q] = bd[q - 1]; bi[q] = bi[q - 1]; --q; }
            bd[q] = d; bi[q] = j;
        }
    }
    for (int k = 0; k < K; ++k) {
        int sel = k;
        if (dil > 1) {
            unsigned h = seed ^ ((unsigned)t * 2654435761u) ^ ((unsigned)k * 40503u);
            h ^= h >> 16; h *= 0x7feb352du; h ^= h >> 15; h *= 0x846ca68bu; h ^= h >> 16;
            sel = (int)(h % (unsigned)Ksel);
        }
        nb[(size_t)t * K + k] = bi[sel];
    }
}

// Neighbor feature rows (bf16, K-padded): A[b,i,k,:] = [x[b,j,:], pos-center, 0pad]
__global__ void gather_feat_kernel(CCE X, CCE POS, CCE CPOS, const int* __restrict__ nb,
                                   unsigned short* __restrict__ A, int np, int n,
                                   int Cx, int K, int Kp, int Btot) {
    size_t t = (size_t)blockIdx.x * blockDim.x + threadIdx.x;
    size_t total = (size_t)Btot * np * K * Kp;
    if (t >= total) return;
    int c = (int)(t % Kp);
    size_t r = t / Kp;
    size_t pi = r / K;                       // b*np + i
    int b = (int)(pi / np);
    int j = nb[r];
    float val = 0.0f;
    if (c < Cx)          val = X[((size_t)b * n + j) * Cx + c];
    else if (c < Cx + 3) val = POS[((size_t)b * n + j) * 3 + (c - Cx)] -
                               CPOS[pi * 3 + (c - Cx)];
    A[t] = f2bf(val);
}

// Max over K neighbors into OUT[p, colofs + c] (channel-concat output, f32)
__global__ void maxk_kernel(CCE Y, float* __restrict__ OUT, int P, int K, int C,
                            int ldOut, int colofs) {
    int t = blockIdx.x * blockDim.x + threadIdx.x;
    if (t >= P * C) return;
    int c = t % C, p = t / C;
    float m = -3.4e38f;
    for (int k = 0; k < K; ++k) m = fmaxf(m, Y[((size_t)p * K + k) * C + c]);
    OUT[(size_t)p * ldOut + colofs + c] = m;
}

__global__ void colmax_kernel(CCE Y, float* __restrict__ OUT, int Btot, int m, int C) {
    int t = blockIdx.x * blockDim.x + threadIdx.x;
    if (t >= Btot * C) return;
    int c = t % C, b = t / C;
    float mx = -3.4e38f;
    for (int i = 0; i < m; ++i) mx = fmaxf(mx, Y[((size_t)b * m + i) * C + c]);
    OUT[t] = mx;
}

// f32 source -> bf16 columns of a staged GEMM A matrix
__global__ void copycols_bf16_kernel(CCE S, unsigned short* __restrict__ D, size_t rows,
                                     int Cs, int ldD, int colofs) {
    size_t t = (size_t)blockIdx.x * blockDim.x + threadIdx.x;
    if (t >= rows * (size_t)Cs) return;
    int c = (int)(t % Cs);
    size_t r = t / Cs;
    D[r * ldD + colofs + c] = f2bf(S[r * Cs + c]);
}

// zero-fill K-pad columns [from, ld)
__global__ void padcols_kernel(unsigned short* __restrict__ D, size_t rows, int ld,
                               int from) {
    int w = ld - from;
    size_t t = (size_t)blockIdx.x * blockDim.x + threadIdx.x;
    if (t >= rows * (size_t)w) return;
    int c = (int)(t % w);
    size_t r = t / w;
    D[r * ld + from + c] = 0;
}

// k<=3 nearest of psrc per pdst point; inverse-squared-distance weights
__global__ void knn3_kernel(CCE pdst, CCE psrc, int n, int m, int k,
                            int* __restrict__ wi, float* __restrict__ ww, int Btot) {
    int t = blockIdx.x * blockDim.x + threadIdx.x;
    if (t >= Btot * n) return;
    int b = t / n;
    float cx = pdst[t * 3], cy = pdst[t * 3 + 1], cz = pdst[t * 3 + 2];
    float bd[3] = {3.4e38f, 3.4e38f, 3.4e38f};
    int   bi[3] = {0, 0, 0};
    const float* P = psrc + (size_t)b * m * 3;
    for (int j = 0; j < m; ++j) {
        float dx = P[j * 3] - cx, dy = P[j * 3 + 1] - cy, dz = P[j * 3 + 2] - cz;
        float d = dx * dx + dy * dy + dz * dz;
        if (d < bd[k - 1]) {
            int q = k - 1;
            while (q > 0 && bd[q - 1] > d) { bd[q] = bd[q - 1]; bi[q] = bi[q - 1]; --q; }
            bd[q] = d; bi[q] = j;
        }
    }
    for (int q = 0; q < k; ++q) {
        wi[(size_t)t * 3 + q] = bi[q];
        ww[(size_t)t * 3 + q] = 1.0f / (bd[q] + 1e-16f);
    }
}

// Weighted kNN interpolation into bf16 A[:, 0..C-1]
__global__ void interp_kernel(CCE xsrc, const int* __restrict__ wi, CCE ww,
                              unsigned short* __restrict__ A, int n, int m, int C,
                              int ldA, int k, int Btot) {
    size_t t = (size_t)blockIdx.x * blockDim.x + threadIdx.x;
    size_t total = (size_t)Btot * n * C;
    if (t >= total) return;
    int c = (int)(t % C);
    size_t r = t / C;
    int b = (int)(r / n);
    float num = 0.0f, den = 0.0f;
    for (int q = 0; q < k; ++q) {
        float w = ww[r * 3 + q];
        int   j = wi[r * 3 + q];
        num += w * xsrc[((size_t)b * m + j) * C + c];
        den += w;
    }
    A[r * ldA + c] = f2bf(num / den);
}

__global__ void logsoftmax_kernel(CCE L, float* __restrict__ O, int rows, int C,
                                  int ldL) {
    int r = blockIdx.x * blockDim.x + threadIdx.x;
    if (r >= rows) return;
    float mx = -3.4e38f;
    for (int c = 0; c < C; ++c) mx = fmaxf(mx, L[(size_t)r * ldL + c]);
    float s = 0.0f;
    for (int c = 0; c < C; ++c) s += expf(L[(size_t)r * ldL + c] - mx);
    float ls = logf(s);
    for (int c = 0; c < C; ++c) O[(size_t)r * C + c] = L[(size_t)r * ldL + c] - mx - ls;
}

__global__ void fill_kernel(float* __restrict__ p, int n, float v) {
    int t = blockIdx.x * blockDim.x + threadIdx.x;
    if (t < n) p[t] = v;
}

// ---------------------------------------------------------------------------
static inline int cdiv(size_t a, size_t b) { return (int)((a + b - 1) / b); }

static void gemm_launch(hipStream_t s, const unsigned short* A, const unsigned short* WT,
                        const float* bias, const float* Add, void* Y,
                        int M, int Kp, int Np, bool relu, bool add, bool out16) {
    long tiles = (long)(M >> 5) * ((Np + 31) >> 5);      // 32x32 blocks per wave
    dim3 g((unsigned)((tiles + 7) / 8)), bl(256);
    if (relu) {
        if (add) {
            if (out16) gemm_wmma_kernel<true, true, true><<<g, bl, 0, s>>>(A, WT, bias, Add, Y, M, Kp, Np);
            else       gemm_wmma_kernel<true, true, false><<<g, bl, 0, s>>>(A, WT, bias, Add, Y, M, Kp, Np);
        } else {
            if (out16) gemm_wmma_kernel<true, false, true><<<g, bl, 0, s>>>(A, WT, bias, Add, Y, M, Kp, Np);
            else       gemm_wmma_kernel<true, false, false><<<g, bl, 0, s>>>(A, WT, bias, Add, Y, M, Kp, Np);
        }
    } else {
        if (add) {
            if (out16) gemm_wmma_kernel<false, true, true><<<g, bl, 0, s>>>(A, WT, bias, Add, Y, M, Kp, Np);
            else       gemm_wmma_kernel<false, true, false><<<g, bl, 0, s>>>(A, WT, bias, Add, Y, M, Kp, Np);
        } else {
            if (out16) gemm_wmma_kernel<false, false, true><<<g, bl, 0, s>>>(A, WT, bias, Add, Y, M, Kp, Np);
            else       gemm_wmma_kernel<false, false, false><<<g, bl, 0, s>>>(A, WT, bias, Add, Y, M, Kp, Np);
        }
    }
}

extern "C" void kernel_launch(void* const* d_in, const int* in_sizes, int n_in,
                              void* d_out, int out_size, void* d_ws, size_t ws_size,
                              hipStream_t stream) {
    (void)in_sizes; (void)n_in; (void)out_size; (void)ws_size;
    const int Bb = 4, Nn = 4096, NP1 = 1024, NP2 = 256;

    const float* x   = (const float*)d_in[0];
    const float* pos = (const float*)d_in[1];

    struct Lin { const float *W, *b; };
    struct Bn  { const float *g, *b, *m, *v; };
    struct Res { Lin l1; Bn n1; Lin l2; Bn n2; Lin ld; Bn nd; };
    int ip = 2;
    auto rdLin = [&](Lin& L) { L.W = (const float*)d_in[ip++]; L.b = (const float*)d_in[ip++]; };
    auto rdBn  = [&](Bn& B)  { B.g = (const float*)d_in[ip++]; B.b = (const float*)d_in[ip++];
                               B.m = (const float*)d_in[ip++]; B.v = (const float*)d_in[ip++]; };
    auto rdRes = [&](Res& R) { rdLin(R.l1); rdBn(R.n1); rdLin(R.l2); rdBn(R.n2);
                               rdLin(R.ld); rdBn(R.nd); };
    Res rsa1[2], rsa2[2], rsa3, rfp3, rfp2, rfp1;
    Res rcs1[2], rcs2[2], rcs3, rcf3, rcf2, rcf1;
    Lin lin1, lin2, lin3;
    rdRes(rsa1[0]); rdRes(rsa1[1]); rdRes(rsa2[0]); rdRes(rsa2[1]); rdRes(rsa3);
    rdRes(rfp3); rdRes(rfp2); rdRes(rfp1);
    rdRes(rcs1[0]); rdRes(rcs1[1]); rdRes(rcs2[0]); rdRes(rcs2[1]); rdRes(rcs3);
    rdRes(rcf3); rdRes(rcf2); rdRes(rcf1);
    rdLin(lin1); rdLin(lin2); rdLin(lin3);

    // ---- workspace arena ----
    char* base = (char*)d_ws; size_t off = 0;
    auto alloc = [&](size_t bytes) -> void* {
        off = (off + 255) & ~(size_t)255;
        void* p = base + off; off += bytes; return p;
    };

    struct FRes { unsigned short *W1, *W2, *Wd; float *b1, *b2, *bd; int cip, cmp, co; };
    auto foldOne = [&](const Lin& L, const Bn* Bp, int ci, int co, int Kp, int Np,
                       unsigned short*& Wo, float*& bo) {
        Wo = (unsigned short*)alloc((size_t)Np * Kp * 2);
        bo = (float*)alloc((size_t)Np * 4);
        fold_kernel<<<cdiv((size_t)Np * Kp, 256), 256, 0, stream>>>(
            L.W, L.b, Bp ? Bp->g : nullptr, Bp ? Bp->b : nullptr,
            Bp ? Bp->m : nullptr, Bp ? Bp->v : nullptr, Wo, bo, ci, co, Kp, Np,
            Bp ? 1 : 0);
    };
    auto foldRes = [&](const Res& R, int ci, int cm, int co) -> FRes {
        FRes F; F.cip = ceil32(ci); F.cmp = ceil32(cm); F.co = co;
        foldOne(R.l1, &R.n1, ci, cm, F.cip, F.cmp, F.W1, F.b1);  // hidden: N-pad->cmp
        foldOne(R.l2, &R.n2, cm, co, F.cmp, F.co,  F.W2, F.b2);
        foldOne(R.ld, &R.nd, ci, co, F.cip, F.co,  F.Wd, F.bd);
        return F;
    };
    FRes Fsa1[2] = {foldRes(rsa1[0], 11, 16, 32),  foldRes(rsa1[1], 11, 16, 32)};
    FRes Fsa2[2] = {foldRes(rsa2[0], 67, 128, 128), foldRes(rsa2[1], 67, 128, 128)};
    FRes Fsa3    =  foldRes(rsa3, 259, 512, 512);
    FRes Ffp3    =  foldRes(rfp3, 768, 512, 256);
    FRes Ffp2    =  foldRes(rfp2, 320, 256, 128);
    FRes Ffp1    =  foldRes(rfp1, 136, 128, 128);
    FRes Fcs1[2] = {foldRes(rcs1[0], 131, 64, 64),  foldRes(rcs1[1], 131, 64, 64)};
    FRes Fcs2[2] = {foldRes(rcs2[0], 131, 128, 128), foldRes(rcs2[1], 131, 128, 128)};
    FRes Fcs3    =  foldRes(rcs3, 259, 512, 512);
    FRes Fcf3    =  foldRes(rcf3, 768, 512, 256);
    FRes Fcf2    =  foldRes(rcf2, 384, 256, 128);
    FRes Fcf1    =  foldRes(rcf1, 256, 128, 128);
    unsigned short *Wl1, *Wl2, *Wl3; float *bl1, *bl2, *bl3;
    foldOne(lin1, nullptr, 128, 128, 128, 128, Wl1, bl1);
    foldOne(lin2, nullptr, 128, 128, 128, 128, Wl2, bl2);
    foldOne(lin3, nullptr, 128, 9,   128, 16,  Wl3, bl3);   // N-pad 9 -> 16

    // ---- persistent activations (f32) ----
    float* s1x  = (float*)alloc((size_t)Bb * NP1 * 64 * 4);
    float* s1p  = (float*)alloc((size_t)Bb * NP1 * 3 * 4);
    float* s2x  = (float*)alloc((size_t)Bb * NP2 * 256 * 4);
    float* s2p  = (float*)alloc((size_t)Bb * NP2 * 3 * 4);
    float* s3x  = (float*)alloc((size_t)Bb * 512 * 4);
    float* gpos = (float*)alloc((size_t)Bb * 3 * 4);
    float* h3   = (float*)alloc((size_t)Bb * NP2 * 256 * 4);
    float* h2   = (float*)alloc((size_t)Bb * NP1 * 128 * 4);
    float* cr   = (float*)alloc((size_t)Bb * Nn * 128 * 4);
    float* c1x  = (float*)alloc((size_t)Bb * NP1 * 128 * 4);
    float* c1p  = (float*)alloc((size_t)Bb * NP1 * 3 * 4);
    float* c2x  = (float*)alloc((size_t)Bb * NP2 * 256 * 4);
    float* c2p  = (float*)alloc((size_t)Bb * NP2 * 3 * 4);
    float* c3x  = (float*)alloc((size_t)Bb * 512 * 4);
    float* ch3  = (float*)alloc((size_t)Bb * NP2 * 256 * 4);
    float* ch2  = (float*)alloc((size_t)Bb * NP1 * 128 * 4);
    float* cr2  = (float*)alloc((size_t)Bb * Nn * 128 * 4);

    // ---- rotating scratch ----
    const size_t MAXROWS = (size_t)Bb * NP1 * 24;                 // 98304
    int*   fpsIdx  = (int*)alloc((size_t)Bb * NP1 * 4);
    float* fpsDist = (float*)alloc((size_t)Bb * Nn * 4);
    int*   nbIdx   = (int*)alloc((size_t)Bb * NP1 * 24 * 4);
    int*   wIdx    = (int*)alloc((size_t)Bb * Nn * 3 * 4);
    float* wW      = (float*)alloc((size_t)Bb * Nn * 3 * 4);
    unsigned short* A16 = (unsigned short*)alloc(MAXROWS * 160 * 2);
    unsigned short* H16 = (unsigned short*)alloc(MAXROWS * 64 * 2);
    float* Tbuf = (float*)alloc(MAXROWS * 64 * 4);
    float* Ybuf = (float*)alloc(MAXROWS * 64 * 4);

    auto runRes = [&](const FRes& F, const unsigned short* Ain, int M, float* Y) {
        gemm_launch(stream, Ain, F.W1, F.b1, nullptr, H16, M, F.cip, F.cmp, true, false, true);
        gemm_launch(stream, Ain, F.Wd, F.bd, nullptr, Tbuf, M, F.cip, F.co, false, false, false);
        gemm_launch(stream, H16, F.W2, F.b2, Tbuf, Y, M, F.cmp, F.co, true, true, false);
    };

    auto runSA = [&](const float* X, const float* POS, int n, int Cx,
                     const FRes* F2, float* OUTX, float* CPOS, unsigned seed) {
        int np = n / 4;
        fps_kernel<<<Bb, 256, 0, stream>>>(POS, n, np, fpsIdx, fpsDist);
        gather_center_kernel<<<cdiv((size_t)Bb * np * 3, 256), 256, 0, stream>>>(
            POS, fpsIdx, CPOS, np, n, Bb);
        const int KS[2] = {8, 24};
        int coTot = F2[0].co + F2[1].co;
        for (int sc = 0; sc < 2; ++sc) {
            int K = KS[sc], Ksel = K * 2;
            knn_kernel<<<cdiv((size_t)Bb * np, 128), 128, 0, stream>>>(
                CPOS, POS, np, n, Ksel, 2, K, seed + 77u * (unsigned)sc, nbIdx, Bb);
            int rows = Bb * np * K, Kp = F2[sc].cip;
            gather_feat_kernel<<<cdiv((size_t)rows * Kp, 256), 256, 0, stream>>>(
                X, POS, CPOS, nbIdx, A16, np, n, Cx, K, Kp, Bb);
            runRes(F2[sc], A16, rows, Ybuf);
            maxk_kernel<<<cdiv((size_t)Bb * np * F2[sc].co, 256), 256, 0, stream>>>(
                Ybuf, OUTX, Bb * np, K, F2[sc].co, coTot, sc * F2[0].co);
        }
    };

    auto runGlobal = [&](const float* X, const float* POS, int m, int C,
                         const FRes& F, float* OUT) {
        int rows = Bb * m, Kp = F.cip;
        copycols_bf16_kernel<<<cdiv((size_t)rows * C, 256), 256, 0, stream>>>(
            X, A16, (size_t)rows, C, Kp, 0);
        copycols_bf16_kernel<<<cdiv((size_t)rows * 3, 256), 256, 0, stream>>>(
            POS, A16, (size_t)rows, 3, Kp, C);
        if (Kp > C + 3)
            padcols_kernel<<<cdiv((size_t)rows * (Kp - C - 3), 256), 256, 0, stream>>>(
                A16, (size_t)rows, Kp, C + 3);
        runRes(F, A16, rows, Ybuf);
        colmax_kernel<<<cdiv((size_t)Bb * F.co, 256), 256, 0, stream>>>(
            Ybuf, OUT, Bb, m, F.co);
    };

    auto runFP = [&](const float* xsrc, const float* psrc, int m, int C,
                     const float* xskip, const float* pdst, int n, int Cs,
                     int k, const FRes& F, float* OUT) {
        knn3_kernel<<<cdiv((size_t)Bb * n, 128), 128, 0, stream>>>(
            pdst, psrc, n, m, k, wIdx, wW, Bb);
        int Kp = F.cip;
        interp_kernel<<<cdiv((size_t)Bb * n * C, 256), 256, 0, stream>>>(
            xsrc, wIdx, wW, A16, n, m, C, Kp, k, Bb);
        copycols_bf16_kernel<<<cdiv((size_t)Bb * n * Cs, 256), 256, 0, stream>>>(
            xskip, A16, (size_t)Bb * n, Cs, Kp, C);
        if (Kp > C + Cs)
            padcols_kernel<<<cdiv((size_t)Bb * n * (Kp - C - Cs), 256), 256, 0, stream>>>(
                A16, (size_t)Bb * n, Kp, C + Cs);
        runRes(F, A16, Bb * n, OUT);
    };

    // ---- forward pass ----
    runSA(x, pos, Nn, 8, Fsa1, s1x, s1p, 0x1001u);
    runSA(s1x, s1p, NP1, 64, Fsa2, s2x, s2p, 0x2002u);
    runGlobal(s2x, s2p, NP2, 256, Fsa3, s3x);
    fill_kernel<<<1, 32, 0, stream>>>(gpos, Bb * 3, 0.0f);
    runFP(s3x, gpos, 1, 512, s2x, s2p, NP2, 256, 1, Ffp3, h3);
    runFP(h3, s2p, NP2, 256, s1x, s1p, NP1, 64, 3, Ffp2, h2);
    runFP(h2, s1p, NP1, 128, x, pos, Nn, 8, 3, Ffp1, cr);

    // coarse-refine loop (LOOP == 1)
    runSA(cr, pos, Nn, 128, Fcs1, c1x, c1p, 0x3003u);
    runSA(c1x, c1p, NP1, 128, Fcs2, c2x, c2p, 0x4004u);
    runGlobal(c2x, c2p, NP2, 256, Fcs3, c3x);
    runFP(c3x, gpos, 1, 512, c2x, c2p, NP2, 256, 1, Fcf3, ch3);
    runFP(ch3, c2p, NP2, 256, c1x, c1p, NP1, 128, 3, Fcf2, ch2);
    runFP(ch2, c1p, NP1, 128, cr, pos, Nn, 128, 3, Fcf1, cr2);

    // classifier head: relu(lin1) -> lin2 -> lin3 -> log_softmax
    int rows = Bb * Nn;
    copycols_bf16_kernel<<<cdiv((size_t)rows * 128, 256), 256, 0, stream>>>(
        cr2, A16, (size_t)rows, 128, 128, 0);
    gemm_launch(stream, A16, Wl1, bl1, nullptr, H16, rows, 128, 128, true, false, true);
    gemm_launch(stream, H16, Wl2, bl2, nullptr, A16, rows, 128, 128, false, false, true);
    gemm_launch(stream, A16, Wl3, bl3, nullptr, Ybuf, rows, 128, 16, false, false, false);
    logsoftmax_kernel<<<cdiv((size_t)rows, 256), 256, 0, stream>>>(
        Ybuf, (float*)d_out, rows, 9, 16);
}
